// Predecessor_87849261073012
// MI455X (gfx1250) — matrix operand order repaired and checked by reading, not compiled, not run on hardware
//
#include <hip/hip_runtime.h>

typedef __attribute__((ext_vector_type(16))) _Float16 v16h;
typedef __attribute__((ext_vector_type(8)))  float    v8f;
typedef __attribute__((ext_vector_type(4)))  float    f4;

#define NODES 8192
#define FEAT  128
#define EDGES 262144

// -------------------------------------------------------------------------
// Kernel 1: P = h[8192x128] x Wt[128x16] via v_wmma_f32_16x16x32_f16.
// Wt col 0 = W[0:128] (src proj), col 1 = W[128:256] (dst proj), cols 2..15 = 0.
// One wave (32 lanes) computes one 16-row tile; K loop = 4 WMMAs.
// -------------------------------------------------------------------------
__global__ void proj_pq_wmma(const float* __restrict__ h,
                             const float* __restrict__ W,
                             float* __restrict__ p,
                             float* __restrict__ q)
{
    const int lane = threadIdx.x & 31;
    const int tile = blockIdx.x * (blockDim.x >> 5) + (threadIdx.x >> 5);
    const int row  = tile * 16 + (lane & 15);   // A-matrix: M = lane%16
    const int col  = lane & 15;                 // B/C/D: N = lane%16
    const int kbA  = (lane >> 4) * 16;          // A: upper half-wave K-base +16
    const int kbB  = (lane >> 4) * 16;          // B: upper half-wave K-base +16

    v8f acc = {};
#pragma unroll
    for (int k0 = 0; k0 < FEAT; k0 += 32) {
        // A tile 16x32 f16: lane holds K = k0+kbA+{0..15}
        v16h a;
        const float* hp = h + (size_t)row * FEAT + k0 + kbA;
#pragma unroll
        for (int e = 0; e < 16; ++e) {
            a[e]     = (_Float16)hp[e];
        }
        // B tile 32x16 f16: lane holds column `col`, K = k0+kbB+{0..15}
        v16h b;
        if (col < 2) {
            const float* wp = W + col * FEAT + k0 + kbB;
#pragma unroll
            for (int e = 0; e < 16; ++e) b[e] = (_Float16)wp[e];
        } else {
#pragma unroll
            for (int e = 0; e < 16; ++e) b[e] = (_Float16)0.0f;
        }
        // (neg_a, A, neg_b, B, c_mod, C, reuse_a, reuse_b)
        acc = __builtin_amdgcn_wmma_f32_16x16x32_f16(
            false, a, false, b, (short)0, acc, false, false);
    }

    // C/D layout: VGPR v holds (M = v + 8*(lane>=16), N = lane%16).
    // Column 0 -> p, column 1 -> q.
    const int rbase = tile * 16 + (lane >> 4) * 8;
    if (col == 0) {
#pragma unroll
        for (int v = 0; v < 8; ++v) p[rbase + v] = acc[v];
    } else if (col == 1) {
#pragma unroll
        for (int v = 0; v < 8; ++v) q[rbase + v] = acc[v];
    }
}

// -------------------------------------------------------------------------
// Kernel 2: fill S (256 MB) with -inf using non-temporal 128-bit stores.
// Dominant cost of the whole problem: pure write bandwidth.
// -------------------------------------------------------------------------
__global__ void fill_neg_inf(f4* __restrict__ out, size_t n4)
{
    const float ninf = -__builtin_huge_valf();
    f4 v = { ninf, ninf, ninf, ninf };
    size_t stride = (size_t)gridDim.x * blockDim.x;
    for (size_t i = (size_t)blockIdx.x * blockDim.x + threadIdx.x; i < n4; i += stride)
        __builtin_nontemporal_store(v, out + i);
}

// -------------------------------------------------------------------------
// Kernel 3: per-edge score = p[src] + q[dst] + ew*W[256] + b, scatter to S.
// Self-edges skipped (left at -inf). Duplicate-edge races match JAX's
// unspecified scatter order.
// -------------------------------------------------------------------------
__global__ void edge_scatter(const int* __restrict__ src,
                             const int* __restrict__ dst,
                             const float* __restrict__ ew,
                             const float* __restrict__ W,
                             const float* __restrict__ b,
                             const float* __restrict__ p,
                             const float* __restrict__ q,
                             float* __restrict__ S)
{
    int e = blockIdx.x * blockDim.x + threadIdx.x;
    if (e >= EDGES) return;
    int s = src[e];
    int d = dst[e];
    if (s == d) return;
    float score = p[s] + q[d] + ew[e] * W[2 * FEAT] + b[0];
    S[(size_t)s * NODES + d] = score;
}

extern "C" void kernel_launch(void* const* d_in, const int* in_sizes, int n_in,
                              void* d_out, int out_size, void* d_ws, size_t ws_size,
                              hipStream_t stream)
{
    const float* h   = (const float*)d_in[0];   // [8192,128]
    const float* ew  = (const float*)d_in[1];   // [262144]
    const float* W   = (const float*)d_in[2];   // [1,257]
    const float* b   = (const float*)d_in[3];   // [1]
    const int*   src = (const int*)d_in[4];     // [262144]
    const int*   dst = (const int*)d_in[5];     // [262144]
    float* S = (float*)d_out;                   // [8192,8192]

    float* p = (float*)d_ws;                    // [8192]
    float* q = p + NODES;                       // [8192] (64 KB total scratch)

    // 1) WMMA projection: 512 row-tiles, 4 waves per 128-thread block.
    proj_pq_wmma<<<NODES / 16 / 4, 128, 0, stream>>>(h, W, p, q);

    // 2) -inf fill: 16,777,216 float4 stores, 4 per thread.
    size_t n4 = (size_t)NODES * NODES / 4;
    fill_neg_inf<<<16384, 256, 0, stream>>>((f4*)S, n4);

    // 3) Edge scatter (serialized after fill on the same stream).
    edge_scatter<<<EDGES / 256, 256, 0, stream>>>(src, dst, ew, W, b, p, q, S);
}
